// ScannedRNN_59682865545612
// MI455X (gfx1250) — matrix operand order repaired and checked by reading, not compiled, not run on hardware
//
#include <hip/hip_runtime.h>
#include <hip/hip_bf16.h>

// GRU scan for MI455X (gfx1250, wave32, WMMA).
// Grid = 2 blocks x 256 threads. Block g owns batch rows [16g,16g+16).
// Wave w (8 per block) owns feature slice d in [16w,16w+16) and computes the
// r/z/n gate columns for that slice. Weights live in VGPRs as bf16 B-tiles;
// h lives bf16 in LDS (WMMA A operand) + f32 in registers (carry).
// Latency design: reset masking applied at h-store time (prefetched flags),
// six independent 4-deep WMMA chains, x pipelined two steps ahead,
// one workgroup barrier per timestep.

typedef __attribute__((ext_vector_type(16))) __bf16 v16bf;
typedef __attribute__((ext_vector_type(8)))  __bf16 v8bf;
typedef __attribute__((ext_vector_type(8)))  float  v8f;

#define T_STEPS 4096
#define BATCH   32
#define DD      128
#define N3D     384

__device__ __forceinline__ float fast_sigmoid(float x) {
    return 1.0f / (1.0f + __expf(-x));
}
__device__ __forceinline__ float fast_tanh(float x) {
    float e = __expf(-2.0f * fabsf(x));
    float t = (1.0f - e) / (1.0f + e);
    return copysignf(t, x);
}
__device__ __forceinline__ v8f splat8(float v) {
    v8f r = {v, v, v, v, v, v, v, v};
    return r;
}

// B-tile [K=32, N=16] bf16 from row-major f32 W[128][384].
// Element e -> K = kb+e (e<8), kb+16+(e-8) (e>=8), kb = k0 + (lane<16 ? 0 : 8).
__device__ __forceinline__ v16bf load_btile(const float* __restrict__ W,
                                            int n0, int k0, int lane) {
    int n  = n0 + (lane & 15);
    int kb = k0 + ((lane & 16) ? 8 : 0);
    v16bf b;
#pragma unroll
    for (int e = 0; e < 16; ++e)
        b[e] = (__bf16)W[(kb + e + (e >= 8 ? 8 : 0)) * N3D + n];
    return b;
}

// A-tile [M=16, K=32] bf16 from LDS row-major [16][128] bf16.
// Lane row m = lane%16. elements 0..7 = K kbase..kbase+7,
// elements 8..15 = K kbase+16..kbase+23, kbase = k0 + (lane<16 ? 0 : 8).
__device__ __forceinline__ v16bf load_atile(const __bf16* buf, int k0, int lane) {
    int m  = lane & 15;
    int k1 = k0 + ((lane & 16) ? 8 : 0);
    v8bf lo = *(const v8bf*)(buf + m * DD + k1);
    v8bf hi = *(const v8bf*)(buf + m * DD + k1 + 16);
    return __builtin_shufflevector(lo, hi, 0, 1, 2, 3, 4, 5, 6, 7,
                                           8, 9, 10, 11, 12, 13, 14, 15);
}

#define WMMA_BF16(A, B, C) \
    __builtin_amdgcn_wmma_f32_16x16x32_bf16(false, (A), false, (B), (short)0, (C), false, false)

__global__ __launch_bounds__(256, 1)
void gru_scan_kernel(const float* __restrict__ x,
                     const unsigned char* __restrict__ resets,
                     const float* __restrict__ Wi,
                     const float* __restrict__ bi,
                     const float* __restrict__ Wh,
                     const float* __restrict__ bhn,
                     float* __restrict__ out) {
    const int g    = blockIdx.x;        // batch half
    const int tid  = threadIdx.x;
    const int wave = tid >> 5;          // 0..7, owns d in [16*wave, 16*wave+16)
    const int lane = tid & 31;
    const int m_hi = (lane >> 4) & 1;

    __shared__ __bf16 xbuf[2][16 * DD]; // x_t slice (bf16), double buffered
    __shared__ __bf16 hbuf[2][16 * DD]; // h_t (bf16, reset-masked), double buffered

    // ---- preload weight B-tiles into registers ----
    const int n_r = 16 * wave;
    const int n_z = 128 + 16 * wave;
    const int n_n = 256 + 16 * wave;
    v16bf Bi_r[4], Bi_z[4], Bi_n[4], Bh_r[4], Bh_z[4], Bh_n[4];
#pragma unroll
    for (int kb = 0; kb < 4; ++kb) {
        Bi_r[kb] = load_btile(Wi, n_r, 32 * kb, lane);
        Bi_z[kb] = load_btile(Wi, n_z, 32 * kb, lane);
        Bi_n[kb] = load_btile(Wi, n_n, 32 * kb, lane);
        Bh_r[kb] = load_btile(Wh, n_r, 32 * kb, lane);
        Bh_z[kb] = load_btile(Wh, n_z, 32 * kb, lane);
        Bh_n[kb] = load_btile(Wh, n_n, 32 * kb, lane);
    }

    const int   dg      = 16 * wave + (lane & 15);
    const float bias_r  = bi[dg];
    const float bias_z  = bi[DD + dg];
    const float bias_xn = bi[2 * DD + dg];
    const float bias_hn = bhn[dg];

    // ---- prologue: stage x_0 to LDS, zero h_0, prefetch x_1 + resets[0] ----
    {
        const float4* xs = (const float4*)(x + (size_t)(16 * g) * DD);
        float4 v0 = xs[2 * tid], v1 = xs[2 * tid + 1];
        v8bf pk = {(__bf16)v0.x, (__bf16)v0.y, (__bf16)v0.z, (__bf16)v0.w,
                   (__bf16)v1.x, (__bf16)v1.y, (__bf16)v1.z, (__bf16)v1.w};
        *((v8bf*)&xbuf[0][8 * tid]) = pk;
        v8bf z8 = {};
        *((v8bf*)&hbuf[0][8 * tid]) = z8;
    }
    float4 xr0, xr1;  // holds x_{t+1} slice during step t
    {
        const float4* xs = (const float4*)(x + ((size_t)BATCH + 16 * g) * DD);
        xr0 = xs[2 * tid];
        xr1 = xs[2 * tid + 1];
    }
    unsigned long long rbits_cur =
        *(const unsigned long long*)(resets + 16 * g + 8 * m_hi);  // resets[0]
    v8f hprev = splat8(0.0f);
    __syncthreads();

#pragma unroll 1
    for (int t = 0; t < T_STEPS; ++t) {
        const int cur = t & 1, nxt = cur ^ 1;

        // stage x_{t+1} (already in regs) to the other buffer; fetch x_{t+2}
        if (t + 1 < T_STEPS) {
            v8bf pk = {(__bf16)xr0.x, (__bf16)xr0.y, (__bf16)xr0.z, (__bf16)xr0.w,
                       (__bf16)xr1.x, (__bf16)xr1.y, (__bf16)xr1.z, (__bf16)xr1.w};
            *((v8bf*)&xbuf[nxt][8 * tid]) = pk;
        }
        if (t + 2 < T_STEPS) {
            const float4* xs =
                (const float4*)(x + ((size_t)(t + 2) * BATCH + 16 * g) * DD);
            xr0 = xs[2 * tid];
            xr1 = xs[2 * tid + 1];
        }
        // prefetch resets[t+1] (used for carry at t+1 AND h-store masking now)
        unsigned long long rbits_next = 0;
        if (t + 1 < T_STEPS)
            rbits_next = *(const unsigned long long*)(
                resets + (size_t)(t + 1) * BATCH + 16 * g + 8 * m_hi);

        // six independent accumulator chains, each 4 WMMAs deep
        v8f acc_rx = splat8(bias_r),  acc_rh = splat8(0.0f);
        v8f acc_zx = splat8(bias_z),  acc_zh = splat8(0.0f);
        v8f acc_xn = splat8(bias_xn), acc_hn = splat8(bias_hn);

#pragma unroll
        for (int kb = 0; kb < 4; ++kb) {
            v16bf a_h = load_atile(&hbuf[cur][0], 32 * kb, lane);  // pre-masked
            v16bf a_x = load_atile(&xbuf[cur][0], 32 * kb, lane);
            acc_rh = WMMA_BF16(a_h, Bh_r[kb], acc_rh);
            acc_zh = WMMA_BF16(a_h, Bh_z[kb], acc_zh);
            acc_hn = WMMA_BF16(a_h, Bh_n[kb], acc_hn);
            acc_rx = WMMA_BF16(a_x, Bi_r[kb], acc_rx);
            acc_zx = WMMA_BF16(a_x, Bi_z[kb], acc_zx);
            acc_xn = WMMA_BF16(a_x, Bi_n[kb], acc_xn);
        }

        // gates + carry (f32)
        v8f hnew;
#pragma unroll
        for (int r = 0; r < 8; ++r) {
            const bool  rst = ((rbits_cur >> (8 * r)) & 0xffull) != 0;
            const float hp  = rst ? 0.0f : hprev[r];
            const float rg  = fast_sigmoid(acc_rx[r] + acc_rh[r]);
            const float zg  = fast_sigmoid(acc_zx[r] + acc_zh[r]);
            const float ng  = fast_tanh(acc_xn[r] + rg * acc_hn[r]);
            hnew[r] = (1.0f - zg) * ng + zg * hp;
        }
        hprev = hnew;

        // h for next step: store reset-masked bf16 (mask off the read path);
        // output: unmasked f32, nontemporal (written once, never re-read)
#pragma unroll
        for (int r = 0; r < 8; ++r) {
            const int  m    = r + 8 * m_hi;
            const bool rstn = ((rbits_next >> (8 * r)) & 0xffull) != 0;
            hbuf[nxt][m * DD + dg] = rstn ? (__bf16)0.0f : (__bf16)hnew[r];
            __builtin_nontemporal_store(
                hnew[r], &out[((size_t)t * BATCH + 16 * g + m) * DD + dg]);
        }
        rbits_cur = rbits_next;
        __syncthreads();
    }
}

extern "C" void kernel_launch(void* const* d_in, const int* in_sizes, int n_in,
                              void* d_out, int out_size, void* d_ws, size_t ws_size,
                              hipStream_t stream) {
    (void)in_sizes; (void)n_in; (void)out_size; (void)d_ws; (void)ws_size;
    const float*         xx  = (const float*)d_in[0];
    const unsigned char* rs  = (const unsigned char*)d_in[1];  // jnp.bool_ = 1 byte
    const float*         Wi  = (const float*)d_in[2];
    const float*         bi  = (const float*)d_in[3];
    const float*         Wh  = (const float*)d_in[4];
    const float*         bhn = (const float*)d_in[5];
    float*               out = (float*)d_out;

    hipLaunchKernelGGL(gru_scan_kernel, dim3(2), dim3(256), 0, stream,
                       xx, rs, Wi, bi, Wh, bhn, out);
}
